// MiniGPT_24369644437870
// MI455X (gfx1250) — matrix operand order repaired and checked by reading, not compiled, not run on hardware
//
#include <hip/hip_runtime.h>
#include <hip/hip_bf16.h>
#include <math.h>
#include <stdint.h>

// ---------------------------------------------------------------------------
// MiniGPT forward for gfx1250 (MI455X): bf16 WMMA everywhere, fp32 accumulate.
// GEMM weight tiles are DMA'd into LDS by the Tensor Data Mover (TDM),
// double-buffered with s_wait_tensorcnt + workgroup barriers.
// ---------------------------------------------------------------------------

typedef __attribute__((ext_vector_type(16))) __bf16       v16bf;
typedef __attribute__((ext_vector_type(8)))  float        v8f;
typedef __attribute__((ext_vector_type(4)))  unsigned int u32x4;
typedef __attribute__((ext_vector_type(8)))  int          i32x8;
typedef __attribute__((ext_vector_type(4)))  int          i32x4;

#define LCFG 8
#define CCH  1024
#define TT   1024
#define BB   4
#define HH   16
#define HD   64
#define MM   (BB * TT)      // 4096 rows
#define FF   (4 * CCH)      // 4096
#define VV   32000

#if __has_builtin(__builtin_amdgcn_tensor_load_to_lds) && \
    __has_builtin(__builtin_amdgcn_s_wait_tensorcnt)
#define USE_TDM 1
#else
#define USE_TDM 0
#endif

// ---------------- WMMA fragment helpers (contiguous K map: K = 16h + j) -----------

__device__ __forceinline__ v8f wmma_bf16(v16bf a, v16bf b, v8f c) {
  return __builtin_amdgcn_wmma_f32_16x16x32_bf16(
      /*neg_a=*/false, a, /*neg_b=*/false, b,
      /*c_mod=*/(short)0, c, /*reuse_a=*/false, /*reuse_b=*/false);
}

// A fragment: 16(M) x 32(K) bf16 from row-major fp32 [.., lda].
// lane L: M = L&15 ; half h = L>>4 covers K in {16h..16h+15}.
__device__ __forceinline__ v16bf load_a_frag_g(const float* __restrict__ A,
                                               int lda, int m0, int k0) {
  const int lane = threadIdx.x & 31;
  const int half = lane >> 4;
  const float* row = A + (size_t)(m0 + (lane & 15)) * lda + k0 + 16 * half;
  float4 f0 = *(const float4*)(row + 0);
  float4 f1 = *(const float4*)(row + 4);
  float4 f2 = *(const float4*)(row + 8);
  float4 f3 = *(const float4*)(row + 12);
  v16bf a;
  a[0]  = (__bf16)f0.x; a[1]  = (__bf16)f0.y; a[2]  = (__bf16)f0.z; a[3]  = (__bf16)f0.w;
  a[4]  = (__bf16)f1.x; a[5]  = (__bf16)f1.y; a[6]  = (__bf16)f1.z; a[7]  = (__bf16)f1.w;
  a[8]  = (__bf16)f2.x; a[9]  = (__bf16)f2.y; a[10] = (__bf16)f2.z; a[11] = (__bf16)f2.w;
  a[12] = (__bf16)f3.x; a[13] = (__bf16)f3.y; a[14] = (__bf16)f3.z; a[15] = (__bf16)f3.w;
  return a;
}

// A fragment from an LDS 16x32 fp32 row-major tile (used by attention).
__device__ __forceinline__ v16bf load_a_frag_lds(const float* t) {
  const int lane = threadIdx.x & 31;
  const int half = lane >> 4;
  const int row  = lane & 15;
  v16bf a;
#pragma unroll
  for (int j = 0; j < 16; ++j) {
    int k = j + 16 * half;   // contiguous K map (matches B)
    a[j] = (__bf16)t[row * 32 + k];
  }
  return a;
}

// B fragment: 32(K) x 16(N) bf16 from row-major fp32 B[K,N] with leading dim ldb.
// lane L: N = L&15 ; element j: K = j + 16*(L>>4).
__device__ __forceinline__ v16bf load_b_frag_g(const float* __restrict__ B,
                                               int ldb, int k0, int n0) {
  const int lane = threadIdx.x & 31;
  const int half = lane >> 4;
  const float* p = B + (size_t)(k0 + 16 * half) * ldb + n0 + (lane & 15);
  v16bf b;
#pragma unroll
  for (int j = 0; j < 16; ++j) b[j] = (__bf16)p[(size_t)j * ldb];
  return b;
}

// B fragment from an LDS 32(K) x 64(N) fp32 row-major tile.
__device__ __forceinline__ v16bf load_b_frag_l(const float* t, int n0) {
  const int lane = threadIdx.x & 31;
  const int half = lane >> 4;
  const float* p = t + (16 * half) * 64 + n0 + (lane & 15);
  v16bf b;
#pragma unroll
  for (int j = 0; j < 16; ++j) b[j] = (__bf16)p[j * 64];
  return b;
}

// B fragment for transposed operand: logical B[K=d][N=s] = Kmat[s][d]
// with Kmat row-major [.., ldk]. Contiguous per lane -> float4 loads.
__device__ __forceinline__ v16bf load_bT_frag_g(const float* __restrict__ Km,
                                                int ldk, int d0, int s0) {
  const int lane = threadIdx.x & 31;
  const int half = lane >> 4;
  const float* row = Km + (size_t)(s0 + (lane & 15)) * ldk + d0 + 16 * half;
  float4 f0 = *(const float4*)(row + 0);
  float4 f1 = *(const float4*)(row + 4);
  float4 f2 = *(const float4*)(row + 8);
  float4 f3 = *(const float4*)(row + 12);
  v16bf b;
  b[0]  = (__bf16)f0.x; b[1]  = (__bf16)f0.y; b[2]  = (__bf16)f0.z; b[3]  = (__bf16)f0.w;
  b[4]  = (__bf16)f1.x; b[5]  = (__bf16)f1.y; b[6]  = (__bf16)f1.z; b[7]  = (__bf16)f1.w;
  b[8]  = (__bf16)f2.x; b[9]  = (__bf16)f2.y; b[10] = (__bf16)f2.z; b[11] = (__bf16)f2.w;
  b[12] = (__bf16)f3.x; b[13] = (__bf16)f3.y; b[14] = (__bf16)f3.z; b[15] = (__bf16)f3.w;
  return b;
}

#if USE_TDM
// Issue a TDM DMA of a 32(K) x 64(N) fp32 tile (row-major, leading dim ldb
// elements) from global into LDS at byte offset lds_off.
// D# layout per CDNA5 ISA (2D tensor; groups 2/3
// zero => VADDR2/3 disabled semantics). This toolchain exposes the 6-arg
// builtin form: (u32x4 g0, i32x8 g1, i32x4, i32x4, i32x8, i32 cpol).
__device__ __forceinline__ void tdm_load_tile_b(const float* gsrc,
                                                unsigned lds_off,
                                                int n_len, int k_len, int ldb) {
  const unsigned long long ga = (unsigned long long)(uintptr_t)gsrc;
  u32x4 g0;
  g0[0] = 1u;                                              // count=1, user mode
  g0[1] = lds_off;                                         // lds_addr (bytes)
  g0[2] = (unsigned)(ga & 0xFFFFFFFFull);                  // global_addr[31:0]
  g0[3] = (unsigned)((ga >> 32) & 0x01FFFFFFull) | (2u << 30); // [56:32] | type=2
  i32x8 g1;
  g1[0] = (int)(2u << 16);                                 // data_size=4B, no mask
  g1[1] = (int)(((unsigned)n_len & 0xFFFFu) << 16);        // tensor_dim0[15:0]
  g1[2] = (int)((((unsigned)n_len >> 16) & 0xFFFFu) |
                (((unsigned)k_len & 0xFFFFu) << 16));      // dim0 hi | dim1 lo
  g1[3] = (int)((((unsigned)k_len >> 16) & 0xFFFFu) |
                (64u << 16));                              // dim1 hi | tile_dim0=64
  g1[4] = 32;                                              // tile_dim1=32, tile_dim2=0
  g1[5] = (int)(unsigned)ldb;                              // tensor_dim0_stride lo32
  g1[6] = 0;                                               // stride hi | dim1_stride lo
  g1[7] = 0;
  i32x4 z4 = {0, 0, 0, 0};
  i32x8 z8 = {0, 0, 0, 0, 0, 0, 0, 0};
  __builtin_amdgcn_tensor_load_to_lds(g0, g1, z4, z4, z8, 0);
}
#endif

// ---------------------------- embedding ------------------------------------

__global__ void embed_kernel(const int* __restrict__ idx,
                             const float* __restrict__ tok,
                             const float* __restrict__ pos,
                             float* __restrict__ x) {
  const int row = blockIdx.x;          // b*T + t
  const int t   = row & (TT - 1);
  const int tid = idx[row];
  const int c   = threadIdx.x * 4;
  float4 te = *(const float4*)(tok + (size_t)tid * CCH + c);
  float4 pe = *(const float4*)(pos + (size_t)t * CCH + c);
  float4 o;
  o.x = te.x + pe.x; o.y = te.y + pe.y; o.z = te.z + pe.z; o.w = te.w + pe.w;
  *(float4*)(x + (size_t)row * CCH + c) = o;
}

// ---------------------------- layernorm ------------------------------------

__global__ __launch_bounds__(256)
void ln_kernel(const float* __restrict__ x, const float* __restrict__ g,
               const float* __restrict__ b, float* __restrict__ h) {
  __shared__ float red[2][8];
  const int row = blockIdx.x;
  const float* xr = x + (size_t)row * CCH;
  const int c = threadIdx.x * 4;
  float4 v = *(const float4*)(xr + c);
  float s  = v.x + v.y + v.z + v.w;
  float s2 = v.x * v.x + v.y * v.y + v.z * v.z + v.w * v.w;
#pragma unroll
  for (int m = 16; m >= 1; m >>= 1) {
    s  += __shfl_xor(s,  m, 32);
    s2 += __shfl_xor(s2, m, 32);
  }
  const int w = threadIdx.x >> 5, lane = threadIdx.x & 31;
  if (lane == 0) { red[0][w] = s; red[1][w] = s2; }
  __syncthreads();
  float ts = 0.f, ts2 = 0.f;
#pragma unroll
  for (int i = 0; i < 8; ++i) { ts += red[0][i]; ts2 += red[1][i]; }
  const float mean = ts * (1.f / CCH);
  const float var  = ts2 * (1.f / CCH) - mean * mean;
  const float inv  = rsqrtf(var + 1e-5f);
  float4 gg = *(const float4*)(g + c);
  float4 bb = *(const float4*)(b + c);
  float4 o;
  o.x = (v.x - mean) * inv * gg.x + bb.x;
  o.y = (v.y - mean) * inv * gg.y + bb.y;
  o.z = (v.z - mean) * inv * gg.z + bb.z;
  o.w = (v.w - mean) * inv * gg.w + bb.w;
  *(float4*)(h + (size_t)row * CCH + c) = o;
}

// ------------------------------ GEMM ---------------------------------------
// C[M,N] = A[M,K] @ B[K,N]  (+bias +resid, relu) ; per-wave 16x64 tile,
// 8 waves/block -> 128(M) x 64(N) per workgroup.
// B tiles (32x64 fp32, shared by all 8 waves) are DMA'd into LDS by the TDM,
// double-buffered: issue tile i+1, s_wait_tensorcnt(1) for tile i, barrier,
// compute, barrier.
// FLAGS: 1=bias, 2=relu, 4=residual, 8=B stored per-head [H][K][64]

template <int FLAGS>
__global__ __launch_bounds__(256)
void gemm_kernel(const float* __restrict__ A, const float* __restrict__ B,
                 const float* __restrict__ bias, const float* __restrict__ resid,
                 float* __restrict__ C, int M, int N, int K) {
  __shared__ float btile[2][32 * 64];
  const int wave = threadIdx.x >> 5;
  const int lane = threadIdx.x & 31;
  const int half = lane >> 4;
  const int m0 = blockIdx.y * 128 + wave * 16;
  const int n0 = blockIdx.x * 64;

  // Per-block B tile geometry (a 64-wide block tile is exactly one head when
  // FLAGS&8, so the per-head layout reduces to a plain 2D tile with ldb=HD).
  const float* Bt;
  int ldb;
  if (FLAGS & 8) { Bt = B + (size_t)(n0 >> 6) * K * HD; ldb = HD; }
  else           { Bt = B + n0;                          ldb = N;  }

  v8f zero = {};
  v8f acc[4] = {zero, zero, zero, zero};

#if USE_TDM
  const unsigned lds0 = (unsigned)(uintptr_t)(void*)&btile[0][0];
  const unsigned lds1 = (unsigned)(uintptr_t)(void*)&btile[1][0];
  if (wave == 0)
    tdm_load_tile_b(Bt, lds0, (FLAGS & 8) ? HD : N, K, ldb);
#endif

  for (int k0 = 0; k0 < K; k0 += 32) {
    const int cur = (k0 >> 5) & 1;
#if USE_TDM
    if (wave == 0) {
      if (k0 + 32 < K) {
        tdm_load_tile_b(Bt + (size_t)(k0 + 32) * ldb, cur ? lds0 : lds1,
                        (FLAGS & 8) ? HD : N, K, ldb);
        __builtin_amdgcn_s_wait_tensorcnt(1);   // tile k0 complete
      } else {
        __builtin_amdgcn_s_wait_tensorcnt(0);
      }
    }
    __syncthreads();
#endif
    if (k0 + 32 < K)
      __builtin_prefetch(A + (size_t)(m0 + (lane & 15)) * K + k0 + 32 + 8 * half, 0, 1);
    v16bf a = load_a_frag_g(A, K, m0, k0);
#pragma unroll
    for (int i = 0; i < 4; ++i) {
#if USE_TDM
      v16bf bfrag = load_b_frag_l(&btile[cur][0], 16 * i);
#else
      v16bf bfrag = load_b_frag_g(Bt, ldb, k0, 16 * i);
#endif
      acc[i] = wmma_bf16(a, bfrag, acc[i]);
    }
#if USE_TDM
    __syncthreads();   // all waves done reading buf before TDM overwrites it
#endif
  }

#pragma unroll
  for (int i = 0; i < 4; ++i) {
    const int col = n0 + 16 * i + (lane & 15);
    float bv = (FLAGS & 1) ? bias[col] : 0.f;
#pragma unroll
    for (int r = 0; r < 8; ++r) {
      const int rowi = m0 + r + 8 * half;
      float v = acc[i][r] + bv;
      if (FLAGS & 4) v += resid[(size_t)rowi * N + col];
      if (FLAGS & 2) v = fmaxf(v, 0.f);
      C[(size_t)rowi * N + col] = v;
    }
  }
}

// --------------------------- attention -------------------------------------
// One wave per (b, h, 16 query rows): flash-style online softmax.
// q/k/v/o layout: [B*T, C] with column h*64+d  (ld = C).

__global__ __launch_bounds__(32)
void attn_kernel(const float* __restrict__ q, const float* __restrict__ k,
                 const float* __restrict__ v, float* __restrict__ o) {
  __shared__ float pbuf[16 * 32];
  const int lane = threadIdx.x & 31;
  const int half = lane >> 4;
  const int t0 = blockIdx.x * 16;
  const int h  = blockIdx.y;
  const int b  = blockIdx.z;
  const size_t base = (size_t)b * TT * CCH + (size_t)h * HD;
  const float* qp = q + base;
  const float* kp = k + base;
  const float* vp = v + base;
  const float scale = 0.125f;   // HD^-0.5

  v16bf aq0 = load_a_frag_g(qp, CCH, t0, 0);
  v16bf aq1 = load_a_frag_g(qp, CCH, t0, 32);

  v8f zero = {};
  v8f oacc[4] = {zero, zero, zero, zero};
  float mrow[8], lrow[8];
#pragma unroll
  for (int r = 0; r < 8; ++r) { mrow[r] = -__builtin_inff(); lrow[r] = 0.f; }

  for (int s0 = 0; s0 <= t0 + 15; s0 += 32) {
    // scores: S[16, 32] = q(16x64) @ k^T, two 16x16 chunks
    v8f sc[2];
#pragma unroll
    for (int c = 0; c < 2; ++c) {
      v8f z = {};
      v16bf b0 = load_bT_frag_g(kp, CCH, 0,  s0 + 16 * c);
      v16bf b1 = load_bT_frag_g(kp, CCH, 32, s0 + 16 * c);
      z = wmma_bf16(aq0, b0, z);
      z = wmma_bf16(aq1, b1, z);
      sc[c] = z;
    }
    // scale + causal mask + online softmax update; stage P into LDS
#pragma unroll
    for (int r = 0; r < 8; ++r) {
      const int qt = t0 + r + 8 * half;
      float v0 = sc[0][r] * scale;
      float v1 = sc[1][r] * scale;
      if (s0 +      (lane & 15) > qt) v0 = -__builtin_inff();
      if (s0 + 16 + (lane & 15) > qt) v1 = -__builtin_inff();
      float mx = fmaxf(v0, v1);
#pragma unroll
      for (int mk = 8; mk >= 1; mk >>= 1) mx = fmaxf(mx, __shfl_xor(mx, mk, 32));
      const float mnew = fmaxf(mrow[r], mx);
      const float corr = __expf(mrow[r] - mnew);
      const float p0 = __expf(v0 - mnew);
      const float p1 = __expf(v1 - mnew);
      float rs = p0 + p1;
#pragma unroll
      for (int mk = 8; mk >= 1; mk >>= 1) rs += __shfl_xor(rs, mk, 32);
      lrow[r] = lrow[r] * corr + rs;
      mrow[r] = mnew;
#pragma unroll
      for (int i = 0; i < 4; ++i) oacc[i][r] *= corr;
      const int prow = r + 8 * half;
      pbuf[prow * 32 +      (lane & 15)] = p0;
      pbuf[prow * 32 + 16 + (lane & 15)] = p1;
    }
    __syncthreads();   // single wave; orders the LDS transpose round-trip
    // O += P(16x32) @ V(32x64)
    v16bf ap = load_a_frag_lds(pbuf);
#pragma unroll
    for (int i = 0; i < 4; ++i) {
      v16bf bv = load_b_frag_g(vp + (size_t)s0 * CCH, CCH, 0, 16 * i);
      oacc[i] = wmma_bf16(ap, bv, oacc[i]);
    }
    __syncthreads();
  }

#pragma unroll
  for (int r = 0; r < 8; ++r) {
    const float inv = 1.f / lrow[r];
    const int rowi = t0 + r + 8 * half;
#pragma unroll
    for (int i = 0; i < 4; ++i)
      o[base + (size_t)rowi * CCH + 16 * i + (lane & 15)] = oacc[i][r] * inv;
  }
}

// ------------------------------ host ---------------------------------------

extern "C" void kernel_launch(void* const* d_in, const int* in_sizes, int n_in,
                              void* d_out, int out_size, void* d_ws, size_t ws_size,
                              hipStream_t stream) {
  const int*   idx  = (const int*)  d_in[0];
  const float* tok  = (const float*)d_in[1];
  const float* pos  = (const float*)d_in[2];
  const float* wq   = (const float*)d_in[3];
  const float* wk   = (const float*)d_in[4];
  const float* wv   = (const float*)d_in[5];
  const float* wpr  = (const float*)d_in[6];
  const float* bpr  = (const float*)d_in[7];
  const float* ln1g = (const float*)d_in[8];
  const float* ln1b = (const float*)d_in[9];
  const float* ln2g = (const float*)d_in[10];
  const float* ln2b = (const float*)d_in[11];
  const float* w1   = (const float*)d_in[12];
  const float* b1   = (const float*)d_in[13];
  const float* w2   = (const float*)d_in[14];
  const float* b2   = (const float*)d_in[15];
  const float* lnfg = (const float*)d_in[16];
  const float* lnfb = (const float*)d_in[17];
  const float* wlm  = (const float*)d_in[18];
  const float* blm  = (const float*)d_in[19];
  float* out = (float*)d_out;

  const size_t ACT = (size_t)MM * CCH;      // 4096*1024 floats
  float* ws = (float*)d_ws;
  float* x  = ws;
  float* hb = x  + ACT;
  float* qb = hb + ACT;
  float* kb = qb + ACT;
  float* vb = kb + ACT;
  float* ao = vb + ACT;
  float* ff = ao + ACT;                     // 4096*4096 floats

  const dim3 blk256(256);
  const dim3 gemmC(CCH / 64, MM / 128);     // N=1024
  const dim3 gemmF(FF / 64,  MM / 128);     // N=4096
  const dim3 gemmV(VV / 64,  MM / 128);     // N=32000
  const dim3 attnG(TT / 16, HH, BB);

  embed_kernel<<<MM, blk256, 0, stream>>>(idx, tok, pos, x);

  for (int l = 0; l < LCFG; ++l) {
    const size_t wOff  = (size_t)l * HH * CCH * HD;   // == l*C*C
    const size_t pOff  = (size_t)l * CCH * CCH;
    const size_t cOff  = (size_t)l * CCH;
    const size_t fOff  = (size_t)l * FF;
    const size_t w1Off = (size_t)l * CCH * FF;

    ln_kernel<<<MM, blk256, 0, stream>>>(x, ln1g + cOff, ln1b + cOff, hb);

    gemm_kernel<8><<<gemmC, blk256, 0, stream>>>(hb, wq + wOff, nullptr, nullptr,
                                                 qb, MM, CCH, CCH);
    gemm_kernel<8><<<gemmC, blk256, 0, stream>>>(hb, wk + wOff, nullptr, nullptr,
                                                 kb, MM, CCH, CCH);
    gemm_kernel<8><<<gemmC, blk256, 0, stream>>>(hb, wv + wOff, nullptr, nullptr,
                                                 vb, MM, CCH, CCH);

    attn_kernel<<<attnG, dim3(32), 0, stream>>>(qb, kb, vb, ao);

    gemm_kernel<1 | 4><<<gemmC, blk256, 0, stream>>>(ao, wpr + pOff, bpr + cOff,
                                                     x, x, MM, CCH, CCH);

    ln_kernel<<<MM, blk256, 0, stream>>>(x, ln2g + cOff, ln2b + cOff, hb);

    gemm_kernel<1 | 2><<<gemmF, blk256, 0, stream>>>(hb, w1 + w1Off, b1 + fOff,
                                                     nullptr, ff, MM, FF, CCH);
    gemm_kernel<1 | 4><<<gemmC, blk256, 0, stream>>>(ff, w2 + w1Off, b2 + cOff,
                                                     x, x, MM, CCH, FF);
  }

  ln_kernel<<<MM, blk256, 0, stream>>>(x, lnfg, lnfb, hb);
  gemm_kernel<1><<<gemmV, blk256, 0, stream>>>(hb, wlm, blm, nullptr,
                                               out, MM, VV, CCH);
}
